// SSLPretrainModel_25486335935037
// MI455X (gfx1250) — matrix operand order repaired and checked by reading, not compiled, not run on hardware
//
#include <hip/hip_runtime.h>

// ---------------------------------------------------------------------------
// chemprop MPN encoder for MI455X (gfx1250).
// Round 3: bf16 storage for msg/a_msg (halves the dominant HBM streams on a
// memory-bound pipeline), 2 M-tiles per wave (each B load feeds 2 WMMAs),
// template-specialized epilogues.
// ---------------------------------------------------------------------------

typedef __bf16 v16bf __attribute__((ext_vector_type(16)));
typedef float  v8f   __attribute__((ext_vector_type(8)));

static constexpr int N_ATOMS = 100000;
static constexpr int N_BONDS = 200000;
static constexpr int N_EDGES = N_BONDS / 2;
static constexpr int N_MOLS  = 2000;
static constexpr int HID     = 300;
static constexpr int HIDW    = HID / 2;  // packed 2xbf16 words per row
static constexpr int AFD     = 133;      // atom feature dim
static constexpr int BFD     = 147;      // bond feature dim
static constexpr int CATD    = AFD + HID; // 433
static constexpr int EFD     = 14;

// n-tiles / k-steps (K padded to multiple of 32, N to multiple of 16)
static constexpr int NT300 = 19, NT133 = 9, NT14 = 1;
static constexpr int KS147 = 5, KS300 = 10, KS433 = 14;

__device__ __forceinline__ unsigned short f2bf(float f) {
  union { float f; unsigned u; } x; x.f = f;
  unsigned r = x.u + 0x7FFFu + ((x.u >> 16) & 1u);   // round-to-nearest-even
  return (unsigned short)(r >> 16);
}
__device__ __forceinline__ float bf2f(unsigned short h) {
  union { unsigned u; float f; } x; x.u = (unsigned)h << 16; return x.f;
}

union AV { v16bf v; unsigned short e[16]; };

// A-operand fetcher: fuses the graph gathers into the GEMM A-load.
// AMODE 0: plain fp32 row       AMODE 1: bf16 a_msg[b2a] - bf16 msg[b2revb]
// AMODE 2: concat(f_atoms fp32, a_msg bf16)   AMODE 3: fp32 endpoint average
template<int AMODE>
struct ARow {
  const void* p0; const void* p1; int kreal;
  __device__ __forceinline__ float fetch(int k) const {
    if (AMODE == 0) {
      const float* f = (const float*)p0;
      return (k < kreal) ? f[k] : 0.f;
    } else if (AMODE == 1) {
      const unsigned short* a = (const unsigned short*)p0;
      const unsigned short* b = (const unsigned short*)p1;
      return (k < HID) ? (bf2f(a[k]) - bf2f(b[k])) : 0.f;
    } else if (AMODE == 2) {
      const float* f = (const float*)p0;
      const unsigned short* h = (const unsigned short*)p1;
      return (k < AFD) ? f[k] : ((k < CATD) ? bf2f(h[k - AFD]) : 0.f);
    } else {
      const float* x = (const float*)p0;
      const float* y = (const float*)p1;
      return (k < HID) ? 0.5f * (x[k] + y[k]) : 0.f;
    }
  }
};

// ---------------------------------------------------------------------------
// Pack weight [K x N] fp32 -> bf16 in per-lane WMMA B layout:
// flat[((nt*ksteps + s)*32 + lane)*16 + i], col = nt*16 + (lane&15),
// k = s*32 + (lane<16 ? 0 : 8) + i + (i<8 ? 0 : 8)     (ISA 7.12.2, 16-bit B 32x16)
// ---------------------------------------------------------------------------
__global__ void pack_b_kernel(const float* __restrict__ W, unsigned short* __restrict__ out,
                              int kreal, int nreal, int ksteps, int ntiles) {
  int t = blockIdx.x * blockDim.x + threadIdx.x;
  int total = ntiles * ksteps * 512;
  if (t >= total) return;
  int i    = t & 15;
  int lane = (t >> 4) & 31;
  int s    = (t >> 9) % ksteps;
  int nt   = t / (512 * ksteps);
  int col  = nt * 16 + (lane & 15);
  int k    = s * 32 + ((lane >> 4) ? 8 : 0) + i + ((i >= 8) ? 8 : 0);
  float v  = (col < nreal && k < kreal) ? W[(size_t)k * nreal + col] : 0.f;
  out[t] = f2bf(v);
}

// ---------------------------------------------------------------------------
// Generic WMMA GEMM. One wave owns MT 16-row M-tiles (A cached in VGPRs for
// all of them); per n-tile one shared B-register set feeds MT WMMA chains.
// OBF: out0 stored as bf16; OUT1: second relu-copy output (bf16).
// ---------------------------------------------------------------------------
template<int AMODE, int KSTEPS, int MT, bool ADD, bool BIAS, bool RELU, bool OBF, bool OUT1>
__global__ __launch_bounds__(256) void gemm_wmma_kernel(
    const void* __restrict__ A0, const void* __restrict__ A1,
    const int* __restrict__ I0, const int* __restrict__ I1,
    int lda, int kreal,
    const unsigned short* __restrict__ Bp, int ntiles,
    const float* __restrict__ addend, const float* __restrict__ bias,
    void* __restrict__ out0, unsigned short* __restrict__ out1,
    int nreal, int mtiles)
{
  const int wave = threadIdx.x >> 5;
  const int lane = threadIdx.x & 31;
  const int wg   = blockIdx.x * (blockDim.x >> 5) + wave;  // group of MT m-tiles
  if (wg * MT >= mtiles) return;         // whole wave exits -> EXEC all-1s for WMMA
  const int rowlo = lane & 15;
  const int hi    = lane >> 4;
  const int kofs  = hi ? 8 : 0;          // ISA 16-bit A layout: lanes 16-31 hold K+8

  ARow<AMODE> ar[MT];
  AV areg[MT][KSTEPS];
#pragma unroll
  for (int t = 0; t < MT; ++t) {
    const int mtile = wg * MT + t;
    if (mtile < mtiles) {                // wave-uniform guard
      const int m = mtile * 16 + rowlo;
      ar[t].kreal = kreal; ar[t].p0 = nullptr; ar[t].p1 = nullptr;
      if (AMODE == 0) {
        ar[t].p0 = (const float*)A0 + (size_t)m * lda;
      } else if (AMODE == 1) {
        int a = I0[m]; int rb = I1[m];
        ar[t].p0 = (const unsigned short*)A0 + (size_t)a * HID;   // a_msg row (bf16)
        ar[t].p1 = (const unsigned short*)A1 + (size_t)rb * HID;  // msg[b2revb] row (bf16)
      } else if (AMODE == 2) {
        ar[t].p0 = (const float*)A0 + (size_t)m * AFD;            // f_atoms row
        ar[t].p1 = (const unsigned short*)A1 + (size_t)m * HID;   // a_msg row (bf16)
      } else {
        int b = 2 * m;                   // even bond of the reverse pair
        int a1 = I0[b]; int rb = I1[b]; int a2 = I0[rb];
        ar[t].p0 = (const float*)A0 + (size_t)a1 * HID;
        ar[t].p1 = (const float*)A0 + (size_t)a2 * HID;
      }
      // Load + convert the A row (two 8-wide K runs per step).
#pragma unroll
      for (int s = 0; s < KSTEPS; ++s) {
        const int kb = s * 32 + kofs;
#pragma unroll
        for (int i = 0; i < 8; ++i) areg[t][s].e[i]     = f2bf(ar[t].fetch(kb + i));
#pragma unroll
        for (int i = 0; i < 8; ++i) areg[t][s].e[8 + i] = f2bf(ar[t].fetch(kb + 16 + i));
      }
    }
  }

  for (int nt = 0; nt < ntiles; ++nt) {
    // one B-register set per n-tile, shared by all MT tile chains
    const unsigned short* bbase = Bp + (((size_t)nt * KSTEPS * 32 + lane) << 4);
    v16bf breg[KSTEPS];
#pragma unroll
    for (int s = 0; s < KSTEPS; ++s)
      breg[s] = *reinterpret_cast<const v16bf*>(bbase + ((size_t)(s * 32) << 4));

    const int col = nt * 16 + rowlo;
#pragma unroll
    for (int t = 0; t < MT; ++t) {
      const int mtile = wg * MT + t;
      if (mtile < mtiles) {
        v8f acc = {};
#pragma unroll
        for (int s = 0; s < KSTEPS; ++s)
          acc = __builtin_amdgcn_wmma_f32_16x16x32_bf16(
                    false, areg[t][s].v, false, breg[s], (short)0, acc, false, false);

        if (col < nreal) {
          const int rbase = mtile * 16 + hi * 8;   // ISA f32 C/D layout
          union { v8f v; float e[8]; } u; u.v = acc;
          float bb = 0.f;
          if (BIAS) bb = bias[col];
          float addv[8];
          if (ADD) {
#pragma unroll
            for (int r = 0; r < 8; ++r)
              addv[r] = addend[(size_t)(rbase + r) * nreal + col];
          }
#pragma unroll
          for (int r = 0; r < 8; ++r) {
            float v = u.e[r];
            if (ADD)  v += addv[r];
            if (BIAS) v += bb;
            if (RELU) v = fmaxf(v, 0.f);
            const size_t idx = (size_t)(rbase + r) * nreal + col;
            if (OBF) ((unsigned short*)out0)[idx] = f2bf(v);
            else     ((float*)out0)[idx] = v;
            if (OUT1) out1[idx] = f2bf(fmaxf(v, 0.f));
          }
        }
      }
    }
  }
}

// ---------------------------------------------------------------------------
// a_msg[a][h] = sum_j msg[a2b[a][j]][h], bf16 in / bf16 out, processed as
// packed 2xbf16 32-bit words (coalesced over the hidden dim).
// ---------------------------------------------------------------------------
__global__ void gather_sum_kernel(const unsigned* __restrict__ msg, const int* __restrict__ a2b,
                                  unsigned* __restrict__ amsg) {
  size_t t = (size_t)blockIdx.x * blockDim.x + threadIdx.x;
  if (t >= (size_t)N_ATOMS * HIDW) return;
  int a = (int)(t / HIDW), c = (int)(t % HIDW);
  const int* nb = a2b + (size_t)a * 6;
  float s0 = 0.f, s1 = 0.f;
#pragma unroll
  for (int j = 0; j < 6; ++j) {
    unsigned p = msg[(size_t)nb[j] * HIDW + c];
    s0 += bf2f((unsigned short)(p & 0xFFFFu));
    s1 += bf2f((unsigned short)(p >> 16));
  }
  amsg[t] = (unsigned)f2bf(s0) | ((unsigned)f2bf(s1) << 16);
}

__global__ void zero_kernel(float* __restrict__ p, int n) {
  int t = blockIdx.x * blockDim.x + threadIdx.x;
  if (t < n) p[t] = 0.f;
}

__global__ void segsum_kernel(const float* __restrict__ ah, const int* __restrict__ gidx,
                              float* __restrict__ gemb) {
  size_t t = (size_t)blockIdx.x * blockDim.x + threadIdx.x;
  if (t >= (size_t)N_ATOMS * HID) return;
  int a = (int)(t / HID), h = (int)(t % HID);
  atomicAdd(&gemb[(size_t)gidx[a] * HID + h], ah[t]);
}

__global__ void g2_kernel(const float* __restrict__ g1, const float* __restrict__ w,
                          const float* __restrict__ b, float* __restrict__ out, int n) {
  int m = blockIdx.x * blockDim.x + threadIdx.x;
  if (m >= n) return;
  float s = b[0];
  const float* row = g1 + (size_t)m * HID;
  for (int k = 0; k < HID; ++k) s += row[k] * w[k];
  out[m] = s;
}

// ---------------------------------------------------------------------------
extern "C" void kernel_launch(void* const* d_in, const int* in_sizes, int n_in,
                              void* d_out, int out_size, void* d_ws, size_t ws_size,
                              hipStream_t stream) {
  (void)in_sizes; (void)n_in; (void)out_size;
  const float* f_atoms   = (const float*)d_in[0];
  const float* f_bonds   = (const float*)d_in[1];
  const int*   a2b       = (const int*)  d_in[2];
  const int*   b2a       = (const int*)  d_in[3];
  const int*   b2revb    = (const int*)  d_in[4];
  const int*   graph_idx = (const int*)  d_in[5];
  const float* W_i    = (const float*)d_in[6];
  const float* W_h    = (const float*)d_in[7];
  const float* W_o    = (const float*)d_in[8];
  const float* b_o    = (const float*)d_in[9];
  const float* W_node = (const float*)d_in[10];
  const float* b_node = (const float*)d_in[11];
  const float* W_edge = (const float*)d_in[12];
  const float* b_edge = (const float*)d_in[13];
  const float* W_g1   = (const float*)d_in[14];
  const float* b_g1   = (const float*)d_in[15];
  const float* W_g2   = (const float*)d_in[16];
  const float* b_g2   = (const float*)d_in[17];

  // ---- workspace layout (all chunks 256B-aligned by construction) ----
  char* ws = (char*)d_ws;
  size_t off = 0;
  auto alloc = [&](size_t bytes) { char* p = ws + off; off += (bytes + 255) & ~(size_t)255; return p; };
  float*    inp   = (float*)alloc((size_t)N_BONDS * HID * 4);        // fp32 pre-activation
  unsigned* msgA  = (unsigned*)alloc((size_t)N_BONDS * HIDW * 4);    // bf16 messages
  unsigned* msgB  = (unsigned*)alloc((size_t)N_BONDS * HIDW * 4);
  unsigned* amsg  = (unsigned*)alloc((size_t)N_ATOMS * HIDW * 4);    // bf16 atom-gather
  float*    ah    = (float*)alloc((size_t)N_ATOMS * HID * 4);        // fp32 atom hiddens
  float*    gemb  = (float*)alloc((size_t)N_MOLS * HID * 4);
  float*    g1buf = (float*)alloc((size_t)N_MOLS * HID * 4);
  unsigned short* WIp  = (unsigned short*)alloc((size_t)NT300 * KS147 * 512 * 2);
  unsigned short* WHp  = (unsigned short*)alloc((size_t)NT300 * KS300 * 512 * 2);
  unsigned short* WOp  = (unsigned short*)alloc((size_t)NT300 * KS433 * 512 * 2);
  unsigned short* WNp  = (unsigned short*)alloc((size_t)NT133 * KS300 * 512 * 2);
  unsigned short* WEp  = (unsigned short*)alloc((size_t)NT14  * KS300 * 512 * 2);
  unsigned short* WG1p = (unsigned short*)alloc((size_t)NT300 * KS300 * 512 * 2);
  if (off > ws_size) return;  // workspace too small; bail rather than corrupt

  float* out_node  = (float*)d_out;
  float* out_edge  = (float*)d_out + (size_t)N_ATOMS * AFD;
  float* out_graph = (float*)d_out + (size_t)N_ATOMS * AFD + (size_t)N_EDGES * EFD;

  auto packGrid = [](int nt, int ks) { return (nt * ks * 512 + 255) / 256; };
  auto gemmGrid = [](int mtiles, int mt) { return ((mtiles + mt - 1) / mt + 7) / 8; };
  const int elemGrid  = (int)(((size_t)N_ATOMS * HID + 255) / 256);
  const int elemGridW = (int)(((size_t)N_ATOMS * HIDW + 255) / 256);

  // ---- pack all weights to bf16 WMMA-B layout (tiny; stays hot in L2) ----
  pack_b_kernel<<<packGrid(NT300, KS147), 256, 0, stream>>>(W_i,    WIp,  BFD,  HID, KS147, NT300);
  pack_b_kernel<<<packGrid(NT300, KS300), 256, 0, stream>>>(W_h,    WHp,  HID,  HID, KS300, NT300);
  pack_b_kernel<<<packGrid(NT300, KS433), 256, 0, stream>>>(W_o,    WOp,  CATD, HID, KS433, NT300);
  pack_b_kernel<<<packGrid(NT133, KS300), 256, 0, stream>>>(W_node, WNp,  HID,  AFD, KS300, NT133);
  pack_b_kernel<<<packGrid(NT14,  KS300), 256, 0, stream>>>(W_edge, WEp,  HID,  EFD, KS300, NT14);
  pack_b_kernel<<<packGrid(NT300, KS300), 256, 0, stream>>>(W_g1,   WG1p, HID,  HID, KS300, NT300);

  const int mtB = N_BONDS / 16;       // 12500
  const int mtA = N_ATOMS / 16;       // 6250
  const int mtE = N_EDGES / 16;       // 6250
  const int mtG = (N_MOLS + 15) / 16; // 125

  // ---- inp = f_bonds @ W_i (fp32) ; msg = relu(inp) (bf16) ----
  gemm_wmma_kernel<0, KS147, 2, false, false, false, false, true>
      <<<gemmGrid(mtB, 2), 256, 0, stream>>>(
      f_bonds, nullptr, nullptr, nullptr, BFD, BFD,
      WIp, NT300, nullptr, nullptr, inp, (unsigned short*)msgA, HID, mtB);

  // ---- DEPTH-1 = 2 message-passing updates (bf16 msg streams) ----
  unsigned* cur = msgA; unsigned* nxt = msgB;
  for (int d = 0; d < 2; ++d) {
    gather_sum_kernel<<<elemGridW, 256, 0, stream>>>(cur, a2b, amsg);
    gemm_wmma_kernel<1, KS300, 2, true, false, true, true, false>
        <<<gemmGrid(mtB, 2), 256, 0, stream>>>(
        amsg, cur, b2a, b2revb, HID, HID,
        WHp, NT300, inp, nullptr, nxt, nullptr, HID, mtB);
    unsigned* t = cur; cur = nxt; nxt = t;
  }

  // ---- final gather, atom_hiddens = relu(concat(f_atoms, a_msg) @ W_o + b_o) ----
  gather_sum_kernel<<<elemGridW, 256, 0, stream>>>(cur, a2b, amsg);
  gemm_wmma_kernel<2, KS433, 1, false, true, true, false, false>
      <<<gemmGrid(mtA, 1), 256, 0, stream>>>(
      f_atoms, amsg, nullptr, nullptr, 0, CATD,
      WOp, NT300, nullptr, b_o, ah, nullptr, HID, mtA);

  // ---- graph pooling (segment sum) ----
  zero_kernel<<<(N_MOLS * HID + 255) / 256, 256, 0, stream>>>(gemb, N_MOLS * HID);
  segsum_kernel<<<elemGrid, 256, 0, stream>>>(ah, graph_idx, gemb);

  // ---- node head ----
  gemm_wmma_kernel<0, KS300, 2, false, true, false, false, false>
      <<<gemmGrid(mtA, 2), 256, 0, stream>>>(
      ah, nullptr, nullptr, nullptr, HID, HID,
      WNp, NT133, nullptr, b_node, out_node, nullptr, AFD, mtA);

  // ---- edge head (endpoint-average fused into A load) ----
  gemm_wmma_kernel<3, KS300, 2, false, true, false, false, false>
      <<<gemmGrid(mtE, 2), 256, 0, stream>>>(
      ah, nullptr, b2a, b2revb, HID, HID,
      WEp, NT14, nullptr, b_edge, out_edge, nullptr, EFD, mtE);

  // ---- graph head ----
  gemm_wmma_kernel<0, KS300, 2, false, true, true, false, false>
      <<<gemmGrid(mtG, 2), 256, 0, stream>>>(
      gemb, nullptr, nullptr, nullptr, HID, HID,
      WG1p, NT300, nullptr, b_g1, g1buf, nullptr, HID, mtG);
  g2_kernel<<<(N_MOLS + 255) / 256, 256, 0, stream>>>(g1buf, W_g2, b_g2, out_graph, N_MOLS);
}
